// SEGNN_37512244363762
// MI455X (gfx1250) — matrix operand order repaired and checked by reading, not compiled, not run on hardware
//
#include <hip/hip_runtime.h>
#include <hip/hip_bf16.h>
#include <math.h>

#define H 256

typedef __attribute__((ext_vector_type(16))) __bf16 v16bf;
typedef __attribute__((ext_vector_type(8)))  float  v8f;

// ---------- helpers ----------

__device__ __forceinline__ float fast_tanh(float x) {
#if __has_builtin(__builtin_amdgcn_tanhf)
  return __builtin_amdgcn_tanhf(x);   // native V_TANH_F32 on CDNA5
#else
  return tanhf(x);
#endif
}

// order-preserving float<->uint encoding so atomicMax(uint) == float max.
// key 0 is the minimum element.
__device__ __forceinline__ unsigned encf(float f) {
  unsigned u = __builtin_bit_cast(unsigned, f);
  return (u & 0x80000000u) ? ~u : (u | 0x80000000u);
}
__device__ __forceinline__ float decf(unsigned k) {
  unsigned u = (k & 0x80000000u) ? (k & 0x7fffffffu) : ~k;
  return __builtin_bit_cast(float, u);
}

__device__ __forceinline__ float dot4(float4 a, float4 b) {
  return a.x * b.x + a.y * b.y + a.z * b.z + a.w * b.w;
}
__device__ __forceinline__ float dot4m(float4 a, float4 x, float4 b) {
  return a.x * x.x * b.x + a.y * x.y * b.y + a.z * x.z * b.z + a.w * x.w * b.w;
}

// A fragment straight from a row-major f32 matrix (used only for the small
// rel GEMM). Per ISA 7.12.2 16-bit A 16x32 layout: lane L -> row M = m0+L%16,
// half = L/16; VGPR j(0..3): K = k0+half*8+{2j,2j+1}; VGPR 4+j: K += 16.
__device__ __forceinline__ v16bf load_a_f32(const float* __restrict__ A, int rows,
                                            int m0, int k0, int lane) {
  int m = m0 + (lane & 15);
  int half = lane >> 4;
  v16bf a;
  if (m < rows) {
    const float* row = A + (long)m * H;
#pragma unroll
    for (int j = 0; j < 4; ++j) {
      int k = k0 + half * 8 + 2 * j;
      a[2 * j]     = (__bf16)row[k];
      a[2 * j + 1] = (__bf16)row[k + 1];
      a[8 + 2 * j] = (__bf16)row[k + 16];
      a[9 + 2 * j] = (__bf16)row[k + 17];
    }
  } else {
#pragma unroll
    for (int j = 0; j < 16; ++j) a[j] = (__bf16)0.0f;
  }
  return a;
}

// ---------- kernels ----------

__global__ void zero_kernel(uint4* __restrict__ p, long n4, unsigned* __restrict__ base,
                            long nword) {
  long i = (long)blockIdx.x * blockDim.x + threadIdx.x;
  long st = (long)gridDim.x * blockDim.x;
  uint4 z = {0u, 0u, 0u, 0u};
  for (long j = i; j < n4; j += st) p[j] = z;
  if (i == 0) for (long j = n4 * 4; j < nword; ++j) base[j] = 0u;  // tail
}

// one wave per edge: three H-dot-products + per-dst running max.
// each lane owns 8 contiguous floats (2 x float4) of each H-vector.
__global__ void score_kernel(const float* __restrict__ ent, const float* __restrict__ rel,
                             const int* __restrict__ src, const int* __restrict__ dst,
                             const int* __restrict__ relid, const int* __restrict__ nodeid,
                             float* __restrict__ sE, float* __restrict__ sN,
                             float* __restrict__ sC,
                             unsigned* __restrict__ mxE, unsigned* __restrict__ mxN,
                             unsigned* __restrict__ mxC, int nE) {
  int e = blockIdx.x * (blockDim.x >> 5) + (threadIdx.x >> 5);
  if (e >= nE) return;
  int lane = threadIdx.x & 31;
  int dn = dst[e];
  int s  = nodeid[src[e]];
  int d  = nodeid[dn];
  int r  = relid[e];
  const float4* hs4 = (const float4*)(ent + (long)s * H);
  const float4* hd4 = (const float4*)(ent + (long)d * H);
  const float4* ev4 = (const float4*)(rel + (long)r * H);
  float pe = 0.f, pn = 0.f, pc = 0.f;
#pragma unroll
  for (int q = 0; q < 2; ++q) {
    float4 a = hs4[2 * lane + q];
    float4 b = hd4[2 * lane + q];
    float4 x = ev4[2 * lane + q];
    pe += dot4(x, b);
    pn += dot4(a, b);
    pc += dot4m(a, x, b);
  }
#pragma unroll
  for (int off = 16; off > 0; off >>= 1) {
    pe += __shfl_xor(pe, off, 32);
    pn += __shfl_xor(pn, off, 32);
    pc += __shfl_xor(pc, off, 32);
  }
  if (lane == 0) {
    sE[e] = pe; sN[e] = pn; sC[e] = pc;
    atomicMax(&mxE[dn], encf(pe));
    atomicMax(&mxN[dn], encf(pn));
    atomicMax(&mxC[dn], encf(pc));
  }
}

// one thread per edge: exp(s - max) and per-dst sums
__global__ void expsum_kernel(const int* __restrict__ dst,
                              float* __restrict__ sE, float* __restrict__ sN,
                              float* __restrict__ sC,
                              const unsigned* __restrict__ mxE,
                              const unsigned* __restrict__ mxN,
                              const unsigned* __restrict__ mxC,
                              float* __restrict__ sumE, float* __restrict__ sumN,
                              float* __restrict__ sumC, int nE) {
  int e = blockIdx.x * blockDim.x + threadIdx.x;
  if (e >= nE) return;
  int d = dst[e];
  float xE = __expf(sE[e] - decf(mxE[d]));
  float xN = __expf(sN[e] - decf(mxN[d]));
  float xC = __expf(sC[e] - decf(mxC[d]));
  sE[e] = xE; sN[e] = xN; sC[e] = xC;
  unsafeAtomicAdd(&sumE[d], xE);
  unsafeAtomicAdd(&sumN[d], xN);
  unsafeAtomicAdd(&sumC[d], xC);
}

// one wave per edge: weighted scatter-add into the three [N,H] accumulators.
// b128 gathers; the 24 per-lane f32 atomic adds are inherent to the scatter.
__global__ void scatter_kernel(const float* __restrict__ ent, const float* __restrict__ rel,
                               const int* __restrict__ src, const int* __restrict__ dst,
                               const int* __restrict__ relid, const int* __restrict__ nodeid,
                               const float* __restrict__ sE, const float* __restrict__ sN,
                               const float* __restrict__ sC,
                               const float* __restrict__ sumE, const float* __restrict__ sumN,
                               const float* __restrict__ sumC,
                               float* __restrict__ accE, float* __restrict__ accN,
                               float* __restrict__ accC, int nE) {
  int e = blockIdx.x * (blockDim.x >> 5) + (threadIdx.x >> 5);
  if (e >= nE) return;
  int lane = threadIdx.x & 31;
  int dn = dst[e];
  int s  = nodeid[src[e]];
  int r  = relid[e];
  float aE = sE[e] / sumE[dn];
  float aN = sN[e] / sumN[dn];
  float aC = sC[e] / sumC[dn];
  const float4* hs4 = (const float4*)(ent + (long)s * H);
  const float4* ev4 = (const float4*)(rel + (long)r * H);
  long base = (long)dn * H + lane * 8;
#pragma unroll
  for (int q = 0; q < 2; ++q) {
    float4 a = hs4[2 * lane + q];
    float4 x = ev4[2 * lane + q];
    float av[4] = {a.x, a.y, a.z, a.w};
    float xv[4] = {x.x, x.y, x.z, x.w};
    long b = base + 4 * q;
#pragma unroll
    for (int i = 0; i < 4; ++i) {
      unsafeAtomicAdd(&accE[b + i], xv[i] * aE);
      unsafeAtomicAdd(&accN[b + i], av[i] * aN);
      unsafeAtomicAdd(&accC[b + i], av[i] * xv[i] * aC);
    }
  }
}

// convert+swizzle f32 accumulators into pre-packed bf16 A-fragments.
// record (mt,kb,lane) = 16 bf16 (32B) laid out exactly as WMMA wants them.
// per-matrix layout: elem offset ((mt*8+kb)*32 + lane)*16 ; one wave per (mt,kb).
__global__ void aconv_kernel(const float* __restrict__ accE, const float* __restrict__ accN,
                             const float* __restrict__ accC,
                             unsigned short* __restrict__ abf, int n) {
  int mtc = (n + 15) >> 4;
  int wid = blockIdx.x * (blockDim.x >> 5) + (threadIdx.x >> 5);
  int lane = threadIdx.x & 31;
  if (wid >= mtc * 8) return;
  int kb = wid & 7;
  int mat = blockIdx.y;
  const float* A = (mat == 0) ? accE : (mat == 1) ? accN : accC;
  int m = ((wid >> 3) << 4) + (lane & 15);
  int half = lane >> 4;
  v16bf a;
  if (m < n) {
    const float* row = A + (long)m * H;
#pragma unroll
    for (int j = 0; j < 4; ++j) {
      int k = (kb << 5) + half * 8 + 2 * j;
      a[2 * j]     = (__bf16)row[k];
      a[2 * j + 1] = (__bf16)row[k + 1];
      a[8 + 2 * j] = (__bf16)row[k + 16];
      a[9 + 2 * j] = (__bf16)row[k + 17];
    }
  } else {
#pragma unroll
    for (int j = 0; j < 16; ++j) a[j] = (__bf16)0.0f;
  }
  *(v16bf*)(abf + (long)mat * mtc * 4096 + ((long)wid * 32 + lane) * 16) = a;
}

// convert+swizzle the four 256x256 f32 weights into bf16 B-fragments.
// per-matrix layout: elem offset ((kb*16+nt)*32 + lane)*16 ; one wave per (mat,kb,nt).
__global__ void wconv_kernel(const float* __restrict__ wE, const float* __restrict__ wN,
                             const float* __restrict__ wC, const float* __restrict__ wR,
                             unsigned short* __restrict__ bwf) {
  int wid = blockIdx.x * (blockDim.x >> 5) + (threadIdx.x >> 5);
  int lane = threadIdx.x & 31;
  if (wid >= 512) return;
  int mat = wid >> 7;
  int kb  = (wid >> 4) & 7;
  int nt  = wid & 15;
  const float* W = (mat == 0) ? wE : (mat == 1) ? wN : (mat == 2) ? wC : wR;
  int ncol = (nt << 4) + (lane & 15);
  int half = lane >> 4;
  v16bf b;
#pragma unroll
  for (int j = 0; j < 8; ++j) {
    int k = (kb << 5) + (j >> 2) * 16 + half * 8 + 2 * (j & 3);
    b[2 * j]     = (__bf16)W[(long)k * H + ncol];
    b[2 * j + 1] = (__bf16)W[(long)(k + 1) * H + ncol];
  }
  *(v16bf*)(bwf + (long)mat * 65536 + (((kb * 16 + nt) * 32 + lane) * 16)) = b;
}

// out_ent = ent + tanh(accE@wE)+tanh(accN@wN)+tanh(accC@wC).
// one 512-thread block per 16-row m-tile; A-frags (3 mats x 8KB) staged into
// LDS via async-to-LDS; wave nt computes n-tile nt; WMMA loop is pure
// ds_load + global b128 + v_wmma.
__global__ void __launch_bounds__(512) gemm_ent(const unsigned short* __restrict__ abf,
                                                const unsigned short* __restrict__ bwf,
                                                const float* __restrict__ ent,
                                                float* __restrict__ out, int n) {
  __shared__ __align__(32) unsigned short smem[3 * 4096];
  int mtc = (n + 15) >> 4;
  int mt = blockIdx.x;
  int t = threadIdx.x;
  int lane = t & 31;
  int nt = t >> 5;

  // stage 3 x 8KB of A-fragments into LDS (one b128 per thread per matrix)
#pragma unroll
  for (int mat = 0; mat < 3; ++mat) {
    const unsigned short* g = abf + (long)mat * mtc * 4096 + (long)mt * 4096 + t * 8;
    unsigned lds = (unsigned)(unsigned long long)&smem[mat * 4096 + t * 8];
    asm volatile("global_load_async_to_lds_b128 %0, %1, off"
                 :: "v"(lds), "v"(g) : "memory");
  }
  asm volatile("s_wait_asynccnt 0x0" ::: "memory");
  __syncthreads();

  float o[8] = {0.f, 0.f, 0.f, 0.f, 0.f, 0.f, 0.f, 0.f};
#pragma unroll
  for (int mat = 0; mat < 3; ++mat) {
    v8f c = {};
#pragma unroll
    for (int kb = 0; kb < 8; ++kb) {
      v16bf a = *(const v16bf*)&smem[mat * 4096 + kb * 512 + lane * 16];
      v16bf b = *(const v16bf*)(bwf + (long)mat * 65536 + ((kb * 16 + nt) * 32 + lane) * 16);
      c = __builtin_amdgcn_wmma_f32_16x16x32_bf16(false, a, false, b,
                                                  (short)0, c, false, false);
    }
#pragma unroll
    for (int r = 0; r < 8; ++r) o[r] += fast_tanh(c[r]);
  }

  int half = lane >> 4;
  int col = (nt << 4) + (lane & 15);
#pragma unroll
  for (int r = 0; r < 8; ++r) {
    long row = (long)(mt << 4) + r + 8 * half;
    if (row < n) {
      long idx = row * H + col;
      out[idx] = ent[idx] + o[r];
    }
  }
}

// out_rel = rel_emb @ rel_w; small (1000x256): one wave per 16x16 tile,
// A converted on the fly, B from pre-swizzled bf16 fragments (matrix slot 3).
__global__ void gemm_rel(const float* __restrict__ rel, const unsigned short* __restrict__ bwf,
                         float* __restrict__ out, int rows) {
  int wid  = (blockIdx.x * blockDim.x + threadIdx.x) >> 5;
  int lane = threadIdx.x & 31;
  int mt = wid >> 4;
  int nt = wid & 15;
  int m0 = mt << 4, n0 = nt << 4;
  if (m0 >= rows) return;   // wave-uniform
  v8f c = {};
#pragma unroll
  for (int kb = 0; kb < 8; ++kb) {
    v16bf a = load_a_f32(rel, rows, m0, kb << 5, lane);
    v16bf b = *(const v16bf*)(bwf + 3L * 65536 + ((kb * 16 + nt) * 32 + lane) * 16);
    c = __builtin_amdgcn_wmma_f32_16x16x32_bf16(false, a, false, b,
                                                (short)0, c, false, false);
  }
  int half = lane >> 4;
  int col  = n0 + (lane & 15);
#pragma unroll
  for (int r = 0; r < 8; ++r) {
    int row = m0 + r + 8 * half;
    if (row < rows) out[(long)row * H + col] = c[r];
  }
}

// ---------- launcher ----------

extern "C" void kernel_launch(void* const* d_in, const int* in_sizes, int n_in,
                              void* d_out, int out_size, void* d_ws, size_t ws_size,
                              hipStream_t stream) {
  const float* ent   = (const float*)d_in[0];
  const float* rel   = (const float*)d_in[1];
  const float* wE    = (const float*)d_in[2];
  const float* wN    = (const float*)d_in[3];
  const float* wC    = (const float*)d_in[4];
  const float* wR    = (const float*)d_in[5];
  const int* src     = (const int*)d_in[6];
  const int* dst     = (const int*)d_in[7];
  const int* relid   = (const int*)d_in[8];
  const int* nodeid  = (const int*)d_in[9];

  const int N  = in_sizes[0] / H;   // 100000
  const int NR = in_sizes[1] / H;   // 1000
  const int nE = in_sizes[6];       // 320000
  const int mtc = (N + 15) / 16;    // 6250

  float* outEnt = (float*)d_out;
  float* outRel = outEnt + (long)N * H;

  // ws layout: accE|accN|accC (3*N*H f32), sums (3*N f32), max keys (3*N u32),
  // scores (3*E f32), bf16 A-fragments (3 * mtc*4096 u16), bf16 B-fragments
  // (4 * 65536 u16). acc..mx contiguous so one zero pass initializes all.
  float* accE = (float*)d_ws;
  float* accN = accE + (long)N * H;
  float* accC = accN + (long)N * H;
  float* sumE = accC + (long)N * H;
  float* sumN = sumE + N;
  float* sumC = sumN + N;
  unsigned* mxE = (unsigned*)(sumC + N);
  unsigned* mxN = mxE + N;
  unsigned* mxC = mxN + N;
  float* sE = (float*)(mxC + N);
  float* sN = sE + nE;
  float* sC = sN + nE;
  unsigned short* abf = (unsigned short*)(sC + nE);
  unsigned short* bwf = abf + 3L * mtc * 4096;

  long nzero = 3L * N * H + 6L * N;
  zero_kernel<<<2048, 256, 0, stream>>>((uint4*)d_ws, nzero / 4,
                                        (unsigned*)d_ws, nzero);

  int edgeBlocks = (nE + 7) / 8;    // 8 waves (edges) per 256-thread block
  score_kernel<<<edgeBlocks, 256, 0, stream>>>(ent, rel, src, dst, relid, nodeid,
                                               sE, sN, sC, mxE, mxN, mxC, nE);

  expsum_kernel<<<(nE + 255) / 256, 256, 0, stream>>>(dst, sE, sN, sC,
                                                      mxE, mxN, mxC,
                                                      sumE, sumN, sumC, nE);

  scatter_kernel<<<edgeBlocks, 256, 0, stream>>>(ent, rel, src, dst, relid, nodeid,
                                                 sE, sN, sC, sumE, sumN, sumC,
                                                 accE, accN, accC, nE);

  wconv_kernel<<<64, 256, 0, stream>>>(wE, wN, wC, wR, bwf);

  aconv_kernel<<<dim3(mtc, 3), 256, 0, stream>>>(accE, accN, accC, abf, N);

  gemm_ent<<<mtc, 512, 0, stream>>>(abf, bwf, ent, outEnt, N);

  int mtR = (NR + 15) / 16;
  gemm_rel<<<(mtR * 16 + 7) / 8, 256, 0, stream>>>(rel, bwf, outRel, NR);
}
